// POWERec_36850819399941
// MI455X (gfx1250) — matrix operand order repaired and checked by reading, not compiled, not run on hardware
//
#include <hip/hip_runtime.h>
#include <math.h>

#define NUM_USER 50000
#define NUM_ITEM 50000
#define N_NODES_TOT (NUM_USER + NUM_ITEM)
#define DIM_E 64
#define N_LAYERS 4

typedef __attribute__((ext_vector_type(8)))  float    v8f;
typedef __attribute__((ext_vector_type(8)))  unsigned v8u;
typedef __attribute__((ext_vector_type(16))) __bf16   v16bf;

// fp32 -> bf16 round-to-nearest-even (used in one-shot weight relayout)
__device__ __forceinline__ __bf16 f2bf(float f) {
  unsigned u = __builtin_bit_cast(unsigned, f);
  u = (u + 0x7fffu + ((u >> 16) & 1u)) >> 16;
  unsigned short s = (unsigned short)u;
  return __builtin_bit_cast(__bf16, s);
}

// Pack two fp32 -> bf16x2 dword with round-half-up: one v_add per float,
// one v_perm_b32 to select the two high halves (lo -> bits[15:0]).
__device__ __forceinline__ unsigned pack2_bf16(float lo, float hi) {
  unsigned ulo = __builtin_bit_cast(unsigned, lo) + 0x8000u;
  unsigned uhi = __builtin_bit_cast(unsigned, hi) + 0x8000u;
  return __builtin_amdgcn_perm(uhi, ulo, 0x07060302u);
}

// Load one A fragment (16-bit 16x32 WMMA A layout) from a row-major fp32 row.
// lane group g = lane>>4:
//   elements 0..7  = K k0 + g*8 + e ; elements 8..15 = K k0 + 16 + g*8 + (e-8)
__device__ __forceinline__ v16bf load_conv_A(const float* __restrict__ arow,
                                             int k0, int g) {
  const float4* p0 = (const float4*)(arow + k0 + g * 8);
  const float4* p1 = (const float4*)(arow + k0 + 16 + g * 8);
  float4 f0 = p0[0], f1 = p0[1];
  float4 f2 = p1[0], f3 = p1[1];
  v8u P;
  P[0] = pack2_bf16(f0.x, f0.y);
  P[1] = pack2_bf16(f0.z, f0.w);
  P[2] = pack2_bf16(f1.x, f1.y);
  P[3] = pack2_bf16(f1.z, f1.w);
  P[4] = pack2_bf16(f2.x, f2.y);
  P[5] = pack2_bf16(f2.z, f2.w);
  P[6] = pack2_bf16(f3.x, f3.y);
  P[7] = pack2_bf16(f3.z, f3.w);
  return __builtin_bit_cast(v16bf, P);
}

// ---------------------------------------------------------------------------
// Relayout W [K,64] fp32 (row-major) into per-lane WMMA B fragments (bf16).
//   lane l: column n = nt*16 + (l&15); K = c*32 + (e<8 ? (l>>4)*8 + e : 16 + (l>>4)*8 + (e-8)), e = 0..15
// Stored as Wb[((c*4 + nt)*32 + l)*16 + e]  -> each lane reads 32B contiguous.
// ---------------------------------------------------------------------------
__global__ void prep_w(const float* __restrict__ W, __bf16* __restrict__ Wb, int K) {
  unsigned t = blockIdx.x * blockDim.x + threadIdx.x;
  if (t >= (unsigned)(K * 64)) return;
  int e  = t & 15;
  int l  = (t >> 4) & 31;
  int nt = (t >> 9) & 3;
  int c  = t >> 11;
  int k  = c * 32 + ((e < 8) ? ((l >> 4) * 8 + e) : (16 + (l >> 4) * 8 + (e - 8)));
  int n  = nt * 16 + (l & 15);
  Wb[t] = f2bf(W[(size_t)k * 64 + n]);
}

// ---------------------------------------------------------------------------
// Users: ego = user_emb + sum(prompt); also init x and the output accumulator.
// outp is pre-offset to (float*)d_out + mod*64 ; row stride 192.
// ---------------------------------------------------------------------------
__global__ void build_users(const float* __restrict__ uemb,
                            const float* __restrict__ prompt,
                            float* __restrict__ ego, float* __restrict__ x,
                            float* __restrict__ outp) {
  unsigned t = blockIdx.x * blockDim.x + threadIdx.x;
  if (t >= (unsigned)NUM_USER * 64u) return;
  int d = t & 63;
  unsigned n = t >> 6;
  float ps = prompt[d] + prompt[64 + d] + prompt[128 + d] + prompt[192 + d];
  float v = uemb[t] + ps;
  ego[t] = v;
  x[t]   = v;
  outp[(size_t)n * 192 + d] = v;
}

// ---------------------------------------------------------------------------
// Items: ego_items = tanh(feat @ W + b) via v_wmma_f32_16x16x32_bf16.
// One wave handles TWO 16-row M-tiles (32 rows). K loop unrolled by 2 with
// named ping-pong buffers and NO control flow in the body: next-chunk indices
// are clamped (scalar cselect) so loads are always valid — no phis, so the
// register allocator can coalesce load/convert results straight into the
// contiguous WMMA operand tuples (no v_mov rotation).
// Requires K % 64 == 0 (true for 64/4096/384).
// ---------------------------------------------------------------------------
__global__ __launch_bounds__(256) void gemm_tanh_wmma(
    const float* __restrict__ feat, const __bf16* __restrict__ Wb,
    const float* __restrict__ bias, int Mrows, int K,
    float* __restrict__ ego, float* __restrict__ x, float* __restrict__ outp) {
  const int lane   = threadIdx.x & 31;
  const int wid    = threadIdx.x >> 5;
  const int mtiles = (Mrows + 15) >> 4;
  const int mslot  = blockIdx.x * 8 + wid;
  const int mtile0 = mslot * 2;
  if (mtile0 >= mtiles) return;                    // wave-uniform exit
  const bool has2 = (mtile0 + 1) < mtiles;

  const int g  = lane >> 4;
  const int mr = lane & 15;
  const int m0 = mtile0 * 16;

  int r0 = m0 + mr;
  int r1 = m0 + 16 + mr;
  if (r1 >= Mrows) r1 = Mrows - 1;                 // clamp tail (loads only)
  const float* __restrict__ arow0 = feat + (size_t)r0 * K;
  const float* __restrict__ arow1 = feat + (size_t)r1 * K;

  v8f a00 = {}, a01 = {}, a02 = {}, a03 = {};
  v8f a10 = {}, a11 = {}, a12 = {}, a13 = {};
  const int nchunk = K >> 5;                       // even by construction
  const __bf16* bl = Wb + (size_t)lane * 16;       // chunk c at bl + c*2048

  // prologue: chunk 0 -> buffer a, chunk 1 -> buffer b
  v16bf A0a = load_conv_A(arow0, 0, g);
  v16bf A1a = load_conv_A(arow1, 0, g);
  v16bf B0a = *(const v16bf*)(bl + 0);
  v16bf B1a = *(const v16bf*)(bl + 512);
  v16bf B2a = *(const v16bf*)(bl + 1024);
  v16bf B3a = *(const v16bf*)(bl + 1536);
  v16bf A0b = load_conv_A(arow0, 32, g);
  v16bf A1b = load_conv_A(arow1, 32, g);
  v16bf B0b = *(const v16bf*)(bl + 2048);
  v16bf B1b = *(const v16bf*)(bl + 2560);
  v16bf B2b = *(const v16bf*)(bl + 3072);
  v16bf B3b = *(const v16bf*)(bl + 3584);

  for (int c = 0; c < nchunk; c += 2) {
    const int cn0 = (c + 2 < nchunk) ? (c + 2) : (nchunk - 2);  // scalar select
    const int cn1 = (c + 3 < nchunk) ? (c + 3) : (nchunk - 1);
    if (c + 6 < nchunk) {
      __builtin_prefetch(arow0 + ((c + 6) << 5), 0, 1);
      __builtin_prefetch(arow1 + ((c + 6) << 5), 0, 1);
    }

    // ---- phase 0: WMMA chunk c (buffer a), then refill a with chunk cn0
    a00 = __builtin_amdgcn_wmma_f32_16x16x32_bf16(false, A0a, false, B0a, (short)0, a00, false, false);
    a01 = __builtin_amdgcn_wmma_f32_16x16x32_bf16(false, A0a, false, B1a, (short)0, a01, false, false);
    a02 = __builtin_amdgcn_wmma_f32_16x16x32_bf16(false, A0a, false, B2a, (short)0, a02, false, false);
    a03 = __builtin_amdgcn_wmma_f32_16x16x32_bf16(false, A0a, false, B3a, (short)0, a03, false, false);
    a10 = __builtin_amdgcn_wmma_f32_16x16x32_bf16(false, A1a, false, B0a, (short)0, a10, false, false);
    a11 = __builtin_amdgcn_wmma_f32_16x16x32_bf16(false, A1a, false, B1a, (short)0, a11, false, false);
    a12 = __builtin_amdgcn_wmma_f32_16x16x32_bf16(false, A1a, false, B2a, (short)0, a12, false, false);
    a13 = __builtin_amdgcn_wmma_f32_16x16x32_bf16(false, A1a, false, B3a, (short)0, a13, false, false);
    {
      const int k = cn0 << 5;
      A0a = load_conv_A(arow0, k, g);
      A1a = load_conv_A(arow1, k, g);
      const __bf16* bn = bl + (size_t)cn0 * 2048;
      B0a = *(const v16bf*)(bn + 0);
      B1a = *(const v16bf*)(bn + 512);
      B2a = *(const v16bf*)(bn + 1024);
      B3a = *(const v16bf*)(bn + 1536);
    }

    // ---- phase 1: WMMA chunk c+1 (buffer b), then refill b with chunk cn1
    a00 = __builtin_amdgcn_wmma_f32_16x16x32_bf16(false, A0b, false, B0b, (short)0, a00, false, false);
    a01 = __builtin_amdgcn_wmma_f32_16x16x32_bf16(false, A0b, false, B1b, (short)0, a01, false, false);
    a02 = __builtin_amdgcn_wmma_f32_16x16x32_bf16(false, A0b, false, B2b, (short)0, a02, false, false);
    a03 = __builtin_amdgcn_wmma_f32_16x16x32_bf16(false, A0b, false, B3b, (short)0, a03, false, false);
    a10 = __builtin_amdgcn_wmma_f32_16x16x32_bf16(false, A1b, false, B0b, (short)0, a10, false, false);
    a11 = __builtin_amdgcn_wmma_f32_16x16x32_bf16(false, A1b, false, B1b, (short)0, a11, false, false);
    a12 = __builtin_amdgcn_wmma_f32_16x16x32_bf16(false, A1b, false, B2b, (short)0, a12, false, false);
    a13 = __builtin_amdgcn_wmma_f32_16x16x32_bf16(false, A1b, false, B3b, (short)0, a13, false, false);
    {
      const int k = cn1 << 5;
      A0b = load_conv_A(arow0, k, g);
      A1b = load_conv_A(arow1, k, g);
      const __bf16* bn = bl + (size_t)cn1 * 2048;
      B0b = *(const v16bf*)(bn + 0);
      B1b = *(const v16bf*)(bn + 512);
      B2b = *(const v16bf*)(bn + 1024);
      B3b = *(const v16bf*)(bn + 1536);
    }
  }

  // C/D layout: VGPR r -> row m0 + r + 8*(lane>>4), col = ntile*16 + (lane&15)
  const int ncol = lane & 15;
#pragma unroll
  for (int r = 0; r < 8; ++r) {
    int row0 = m0 + r + 8 * g;
    float t0[4] = { a00[r], a01[r], a02[r], a03[r] };
#pragma unroll
    for (int nt = 0; nt < 4; ++nt) {
      int col = nt * 16 + ncol;
      float v = tanhf(t0[nt] + bias[col]);
      ego[(size_t)row0 * 64 + col]   = v;
      x[(size_t)row0 * 64 + col]     = v;
      outp[(size_t)row0 * 192 + col] = v;
    }
  }
  if (has2) {
#pragma unroll
    for (int r = 0; r < 8; ++r) {
      int row1 = m0 + 16 + r + 8 * g;
      float t1[4] = { a10[r], a11[r], a12[r], a13[r] };
#pragma unroll
      for (int nt = 0; nt < 4; ++nt) {
        int col = nt * 16 + ncol;
        float v = tanhf(t1[nt] + bias[col]);
        ego[(size_t)row1 * 64 + col]   = v;
        x[(size_t)row1 * 64 + col]     = v;
        outp[(size_t)row1 * 192 + col] = v;
      }
    }
  }
}

// ---------------------------------------------------------------------------
__global__ void row_norms(const float* __restrict__ ego, float* __restrict__ nrm) {
  unsigned n = blockIdx.x * blockDim.x + threadIdx.x;
  if (n >= (unsigned)N_NODES_TOT) return;
  const float4* p = (const float4*)(ego + (size_t)n * 64);
  float s = 0.f;
#pragma unroll
  for (int i = 0; i < 16; ++i) {
    float4 f = p[i];
    s += f.x * f.x + f.y * f.y + f.z * f.z + f.w * f.w;
  }
  nrm[n] = sqrtf(s);
}

__global__ void zero_f4(float4* __restrict__ p, unsigned n4) {
  unsigned i = blockIdx.x * blockDim.x + threadIdx.x;
  if (i < n4) p[i] = make_float4(0.f, 0.f, 0.f, 0.f);
}

// Edge-parallel SpMM: one wave per edge, 2 dims per lane; x lives in L2 (25.6MB).
__global__ __launch_bounds__(256) void spmm_atomic(
    const int* __restrict__ rows, const int* __restrict__ cols,
    const float* __restrict__ vals, const float* __restrict__ x,
    float* __restrict__ y, int nnz) {
  unsigned tid = blockIdx.x * blockDim.x + threadIdx.x;
  int e    = (int)(tid >> 5);
  int lane = threadIdx.x & 31;
  if (e >= nnz) return;
  int   r = rows[e];
  int   c = cols[e];
  float v = vals[e];
  float2 xv = ((const float2*)(x + (size_t)c * 64))[lane];
  float* yp = y + (size_t)r * 64 + 2 * lane;
  atomicAdd(yp,     v * xv.x);
  atomicAdd(yp + 1, v * xv.y);
}

// Cosine reweight: w = <x,ego> / max(|x||ego|, eps); x *= w (in place); out += x.
__global__ __launch_bounds__(256) void layer_update(
    const float* __restrict__ ego, const float* __restrict__ nrm,
    float* __restrict__ xn, float* __restrict__ outp) {
  unsigned n = blockIdx.x * blockDim.x + threadIdx.x;
  if (n >= (unsigned)N_NODES_TOT) return;
  float4*       xp = (float4*)(xn + (size_t)n * 64);
  const float4* ep = (const float4*)(ego + (size_t)n * 64);
  float dot = 0.f, ss = 0.f;
  float4 xv[16];
#pragma unroll
  for (int i = 0; i < 16; ++i) {
    float4 a = xp[i];
    float4 b = ep[i];
    xv[i] = a;
    dot += a.x * b.x + a.y * b.y + a.z * b.z + a.w * b.w;
    ss  += a.x * a.x + a.y * a.y + a.z * a.z + a.w * a.w;
  }
  float w = dot / fmaxf(sqrtf(ss) * nrm[n], 1e-8f);
  float* op = outp + (size_t)n * 192;
#pragma unroll
  for (int i = 0; i < 16; ++i) {
    float4 a = xv[i];
    a.x *= w; a.y *= w; a.z *= w; a.w *= w;
    xp[i] = a;
    op[4 * i + 0] += a.x;
    op[4 * i + 1] += a.y;
    op[4 * i + 2] += a.z;
    op[4 * i + 3] += a.w;
  }
}

// ---------------------------------------------------------------------------
extern "C" void kernel_launch(void* const* d_in, const int* in_sizes, int n_in,
                              void* d_out, int out_size, void* d_ws, size_t ws_size,
                              hipStream_t stream) {
  (void)n_in; (void)out_size; (void)ws_size;
  const int*   adj_rows  = (const int*)  d_in[0];
  const int*   adj_cols  = (const int*)  d_in[1];
  const float* adj_vals  = (const float*)d_in[2];
  const float* user_emb  = (const float*)d_in[3];
  const float* item_emb  = (const float*)d_in[4];
  const float* v_feat    = (const float*)d_in[5];
  const float* t_feat    = (const float*)d_in[6];
  const float* id_prompt = (const float*)d_in[7];
  const float* v_prompt  = (const float*)d_in[8];
  const float* t_prompt  = (const float*)d_in[9];
  const float* W_id      = (const float*)d_in[10];
  const float* b_id      = (const float*)d_in[11];
  const float* W_v       = (const float*)d_in[12];
  const float* b_v       = (const float*)d_in[13];
  const float* W_t       = (const float*)d_in[14];
  const float* b_t       = (const float*)d_in[15];
  const int nnz = in_sizes[0];

  float* out = (float*)d_out;

  // workspace layout (~78 MB)
  const size_t NE = (size_t)N_NODES_TOT * DIM_E;
  float*  ego = (float*)d_ws;
  float*  xa  = ego + NE;
  float*  xb  = xa + NE;
  float*  nrm = xb + NE;
  __bf16* Wb  = (__bf16*)(nrm + N_NODES_TOT);   // 32B-aligned by construction

  struct Mod { const float* feat; int K; const float* W; const float* b; const float* prompt; };
  Mod mods[3] = {
    { item_emb,   64, W_id, b_id, id_prompt },
    { v_feat,   4096, W_v,  b_v,  v_prompt  },
    { t_feat,    384, W_t,  b_t,  t_prompt  },
  };

  const int mtiles = (NUM_ITEM + 15) / 16;       // 3125
  const int mslots = (mtiles + 1) / 2;           // 1563 (2 tiles per wave)
  const int gblocks = (mslots + 7) / 8;          // 196

  for (int m = 0; m < 3; ++m) {
    const Mod& md = mods[m];

    int wtot = md.K * 64;
    prep_w<<<(wtot + 255) / 256, 256, 0, stream>>>(md.W, Wb, md.K);

    build_users<<<(NUM_USER * 64 + 255) / 256, 256, 0, stream>>>(
        user_emb, md.prompt, ego, xa, out + m * 64);

    gemm_tanh_wmma<<<gblocks, 256, 0, stream>>>(
        md.feat, Wb, md.b, NUM_ITEM, md.K,
        ego + (size_t)NUM_USER * 64,
        xa  + (size_t)NUM_USER * 64,
        out + (size_t)NUM_USER * 192 + m * 64);

    row_norms<<<(N_NODES_TOT + 255) / 256, 256, 0, stream>>>(ego, nrm);

    float* xcur  = xa;
    float* xnext = xb;
    for (int l = 0; l < N_LAYERS; ++l) {
      zero_f4<<<(unsigned)((NE / 4 + 255) / 256), 256, 0, stream>>>(
          (float4*)xnext, (unsigned)(NE / 4));
      spmm_atomic<<<(nnz + 7) / 8, 256, 0, stream>>>(
          adj_rows, adj_cols, adj_vals, xcur, xnext, nnz);
      layer_update<<<(N_NODES_TOT + 255) / 256, 256, 0, stream>>>(
          ego, nrm, xnext, out + m * 64);
      float* t = xcur; xcur = xnext; xnext = t;
    }
  }
}
